// ReGCN_19189913878662
// MI455X (gfx1250) — compile-verified, matrix operand order checked by reading, not run on hardware
//
#include <hip/hip_runtime.h>
#include <math.h>

typedef __bf16 bf16;
typedef __attribute__((ext_vector_type(16))) __bf16 v16bf;
typedef __attribute__((ext_vector_type(8)))  float  v8f;

#define B_  8
#define N_  1024
#define R_  3
#define F_  128
#define H_  128
#define BN  8192   // B_*N_

// ---------- WMMA helpers (CDNA5 bf16 16x16x32, f32 accum) ----------

// Load a 16x32 bf16 A-fragment (or B-fragment from K-major/"transposed" storage).
// Per ISA: lane<16 -> row (lane&15), K sel=0 (K 0..7 in VGPR0-3, K 16..23 in VGPR4-7)
//          lane>=16 -> row (lane&15), K sel=8 (K 8..15, K 24..31)
// Caller passes p = base + row*stride + sel (16B aligned); we read p[0..7], p[16..23].
__device__ __forceinline__ v16bf load_frag(const bf16* p) {
  v16bf f;
  *((int4*)&f)       = *(const int4*)(p);
  *(((int4*)&f) + 1) = *(const int4*)(p + 16);
  return f;
}

__device__ __forceinline__ v8f wmma_bf(v16bf a, v16bf b, v8f c) {
  return __builtin_amdgcn_wmma_f32_16x16x32_bf16(false, a, false, b, (short)0, c, false, false);
}

__device__ __forceinline__ v8f vzero8() {
  v8f z = {0.f,0.f,0.f,0.f,0.f,0.f,0.f,0.f};
  return z;
}

// CDNA5 async global->LDS copy (ASYNCcnt-tracked), 16B per lane.
// lds_off: byte offset in LDS (low 32 bits of a generic LDS pointer).
__device__ __forceinline__ void async_load_b128(unsigned lds_off, const void* gaddr) {
  asm volatile("global_load_async_to_lds_b128 %0, %1, off"
               :: "v"(lds_off), "v"(gaddr) : "memory");
}
__device__ __forceinline__ void wait_asynccnt0() {
  asm volatile("s_wait_asynccnt 0x0" ::: "memory");
}

__device__ __forceinline__ void atomicMaxF(float* addr, float val) {
  unsigned int* a = (unsigned int*)addr;
  unsigned int old = *a;
  while (true) {
    float cur = __uint_as_float(old);
    if (cur >= val) break;
    unsigned int assumed = old;
    old = atomicCAS(a, assumed, __float_as_uint(val));
    if (old == assumed) break;
  }
}

// ---------- 1. token soft-attention:  [B,N,32,F] -> Xb [BN,F] bf16 ----------
__global__ __launch_bounds__(128)
void k_tok(const float* __restrict__ inputs, const float* __restrict__ sa0_w,
           const float* __restrict__ sa0_b, const float* __restrict__ ln0_w,
           const float* __restrict__ ln0_b, bf16* __restrict__ Xb) {
  int bn = blockIdx.x;          // 0..BN-1
  int f  = threadIdx.x;         // 0..127
  __shared__ float sw[32], lw[32], lb[32];
  __shared__ float sb;
  if (threadIdx.x < 32) {
    sw[threadIdx.x] = sa0_w[threadIdx.x];
    lw[threadIdx.x] = ln0_w[threadIdx.x];
    lb[threadIdx.x] = ln0_b[threadIdx.x];
  }
  if (threadIdx.x == 0) sb = sa0_b[0];
  __syncthreads();

  const float* base = inputs + (size_t)bn * 32 * F_ + f;
  float xs[32];
  float dot = 0.f;
  #pragma unroll
  for (int c = 0; c < 32; ++c) { xs[c] = base[c * F_]; dot += xs[c] * sw[c]; }
  float t = tanhf(dot + sb);
  float mx = -3.4e38f;
  float eg[32];
  #pragma unroll
  for (int c = 0; c < 32; ++c) { eg[c] = t * lw[c] + lb[c]; mx = fmaxf(mx, eg[c]); }
  float s = 0.f;
  #pragma unroll
  for (int c = 0; c < 32; ++c) { eg[c] = expf(eg[c] - mx); s += eg[c]; }
  float acc = 0.f;
  #pragma unroll
  for (int c = 0; c < 32; ++c) acc += eg[c] * xs[c];
  acc /= s;
  Xb[(size_t)bn * F_ + f] = (bf16)acc;
}

// ---------- 2. f32 -> bf16 cast (4-wide) ----------
struct alignas(8) BF4 { bf16 a, b, c, d; };
__global__ __launch_bounds__(256)
void k_cast4(const float* __restrict__ in, bf16* __restrict__ out, int n4) {
  int i = blockIdx.x * 256 + threadIdx.x;
  if (i < n4) {
    float4 v = ((const float4*)in)[i];
    BF4 o; o.a = (bf16)v.x; o.b = (bf16)v.y; o.c = (bf16)v.z; o.d = (bf16)v.w;
    ((BF4*)out)[i] = o;
  }
}

// ---------- 3a. basis weights, stored transposed: WT[r][o][i] ----------
__global__ __launch_bounds__(256)
void k_basis(const float* __restrict__ w_rel, const float* __restrict__ w_bases,
             bf16* __restrict__ WT) {
  int r = blockIdx.y;
  int idx = blockIdx.x * 256 + threadIdx.x;   // 16384 per r
  int o = idx >> 7, i = idx & 127;
  float s = 0.f;
  #pragma unroll
  for (int q = 0; q < 3; ++q) s += w_rel[r * 3 + q] * w_bases[((size_t)q * F_ + i) * H_ + o];
  WT[((size_t)r * H_ + o) * F_ + i] = (bf16)s;
}

// ---------- 3b. plain 128x128 transpose+cast: WT[o][i] = W[i][o] ----------
__global__ __launch_bounds__(256)
void k_mat_t(const float* __restrict__ W, bf16* __restrict__ WT) {
  int idx = blockIdx.x * 256 + threadIdx.x;   // 16384
  int o = idx >> 7, i = idx & 127;
  WT[(size_t)o * H_ + i] = (bf16)W[(size_t)i * H_ + o];
}

// ---------- 4. GEMM: C[M,128] = A[M,128](bf16,row-major) x W (WT[o][i] bf16) ----------
// Compile-time output mode: optional bias; row-major bf16 and/or transposed bf16 [128][ldT].
template<bool HAS_BIAS, bool OUT_ROW, bool OUT_T>
__global__ __launch_bounds__(256)
void k_gemm128(const bf16* __restrict__ A, const bf16* __restrict__ WT,
               const float* __restrict__ bias,
               bf16* __restrict__ outRow, bf16* __restrict__ outT, int ldT) {
  int wave = threadIdx.x >> 5, lane = threadIdx.x & 31;
  int rr = lane & 15, sel = (lane < 16) ? 0 : 8;
  int row0 = blockIdx.x * 128 + wave * 16;

  v16bf af[4];
  const bf16* ap = A + (size_t)(row0 + rr) * F_ + sel;
  #pragma unroll
  for (int ks = 0; ks < 4; ++ks) af[ks] = load_frag(ap + ks * 32);

  #pragma unroll
  for (int ct = 0; ct < 8; ++ct) {
    v8f acc = vzero8();
    const bf16* bp = WT + (size_t)(ct * 16 + rr) * F_ + sel;
    #pragma unroll
    for (int ks = 0; ks < 4; ++ks) {
      v16bf bfr = load_frag(bp + ks * 32);
      acc = wmma_bf(af[ks], bfr, acc);
    }
    int ncol = lane & 15, mo = (lane < 16) ? 0 : 8;
    int h = ct * 16 + ncol;
    float bv = HAS_BIAS ? bias[h] : 0.f;
    #pragma unroll
    for (int v = 0; v < 8; ++v) {
      int m = row0 + mo + v;
      float val = acc[v] + bv;
      if (OUT_ROW) outRow[(size_t)m * H_ + h] = (bf16)val;
      if (OUT_T)   outT[(size_t)h * ldT + m] = (bf16)val;
    }
  }
}

// ---------- 5. SpMM: C[b] = sum_r adj[b,r](NxN) x Y_r  with fused epilogue ----------
// adjb: [B][R][N][N] bf16 row-major; YT: [R][H][BN] bf16 (transposed Y).
// adj 32x128 chunks are staged into LDS with CDNA5 async global->LDS copies.
template<bool LAYER1>
__global__ __launch_bounds__(256)
void k_spmm(const bf16* __restrict__ adjb, const bf16* __restrict__ YT,
            const float* __restrict__ bias, const float* __restrict__ mask,
            const float* __restrict__ Xprev, float* __restrict__ outX,
            bf16* __restrict__ outXb) {
  __shared__ bf16 tileA[32 * 128];   // 8 KB staging for adj chunk

  int b  = blockIdx.y;
  int n0 = blockIdx.x * 32;
  int wave = threadIdx.x >> 5, lane = threadIdx.x & 31;
  int rt = wave & 1;          // row tile (16 rows)
  int cg = wave >> 1;         // column group: cols [cg*32, cg*32+32)
  int rr = lane & 15, sel = (lane < 16) ? 0 : 8;

  // staging map: thread -> 16B (8 bf16); two rounds cover 32 rows x 128 cols
  int srow = threadIdx.x >> 4;          // 0..15
  int scol = (threadIdx.x & 15) * 8;    // 0..120
  unsigned lds0 = (unsigned)(size_t)(tileA + srow * 128 + scol);
  unsigned lds1 = (unsigned)(size_t)(tileA + (srow + 16) * 128 + scol);

  v8f acc0 = vzero8(), acc1 = vzero8();
  for (int r = 0; r < R_; ++r) {
    const bf16* abase = adjb + ((size_t)(b * R_ + r) * N_ + n0) * N_;
    const bf16* ybase = YT + (size_t)r * H_ * BN + (size_t)b * N_;
    for (int kc = 0; kc < 8; ++kc) {       // 128-col chunks of adj
      int c0 = kc * 128;
      __syncthreads();                      // previous chunk fully consumed
      async_load_b128(lds0, abase + (size_t)srow * N_ + c0 + scol);
      async_load_b128(lds1, abase + (size_t)(srow + 16) * N_ + c0 + scol);
      wait_asynccnt0();
      __syncthreads();                      // staging visible to all waves
      #pragma unroll
      for (int ks = 0; ks < 4; ++ks) {
        int ml = ks * 32;                   // col within chunk
        int m0 = c0 + ml;                   // global contraction index
        v16bf a  = load_frag(tileA + (rt * 16 + rr) * 128 + ml + sel);
        v16bf b0 = load_frag(ybase + (size_t)(cg * 32 + rr) * BN + m0 + sel);
        acc0 = wmma_bf(a, b0, acc0);
        v16bf b1 = load_frag(ybase + (size_t)(cg * 32 + 16 + rr) * BN + m0 + sel);
        acc1 = wmma_bf(a, b1, acc1);
      }
    }
  }

  int ncol = lane & 15, mo = (lane < 16) ? 0 : 8;
  #pragma unroll
  for (int t = 0; t < 2; ++t) {
    v8f acc = t ? acc1 : acc0;
    int h = cg * 32 + t * 16 + ncol;
    float bv = bias[h];
    #pragma unroll
    for (int v = 0; v < 8; ++v) {
      int m = n0 + rt * 16 + mo + v;
      size_t gi = (size_t)b * N_ + m;
      float val = fmaxf(acc[v] + bv, 0.f) * mask[gi];
      if (LAYER1) {
        val = 0.4f * Xprev[gi * H_ + h] + 0.6f * val;
      } else {
        outX[gi * H_ + h] = val;
      }
      outXb[gi * H_ + h] = (bf16)val;
    }
  }
}

// ---------- 6. attention: softmax(Q K^T / sqrt(H)) V, then column sum & max ----------
// Q,K: [B][N][H] bf16 row-major; VT: [H][BN] bf16.
__global__ __launch_bounds__(256)
void k_attn(const bf16* __restrict__ Qb, const bf16* __restrict__ Kb,
            const bf16* __restrict__ VTb, float* __restrict__ gsum,
            float* __restrict__ gmax) {
  extern __shared__ char smem[];
  float* S = (float*)smem;                                    // [32][1024] f32
  bf16*  P = (bf16*)(smem + (size_t)32 * N_ * sizeof(float)); // [32][1024] bf16

  const float scale = 0.08838834764831845f;  // 1/sqrt(128)
  int b  = blockIdx.y;
  int q0 = blockIdx.x * 32;
  int wave = threadIdx.x >> 5, lane = threadIdx.x & 31;
  int rr = lane & 15, sel = (lane < 16) ? 0 : 8;
  int ncol = lane & 15, mo = (lane < 16) ? 0 : 8;
  int rt = wave & 1;
  int cg = wave >> 1;

  // phase 1: S = scale * Q Kt
  {
    v16bf aq[4];
    const bf16* qp = Qb + ((size_t)(b * N_) + q0 + rt * 16 + rr) * H_ + sel;
    #pragma unroll
    for (int ks = 0; ks < 4; ++ks) aq[ks] = load_frag(qp + ks * 32);
    for (int t = 0; t < 16; ++t) {
      int ct = cg * 16 + t;              // key tile -> keys [ct*16, ct*16+16)
      v8f acc = vzero8();
      const bf16* kp = Kb + ((size_t)(b * N_) + ct * 16 + rr) * H_ + sel;
      #pragma unroll
      for (int ks = 0; ks < 4; ++ks) {
        v16bf bfr = load_frag(kp + ks * 32);
        acc = wmma_bf(aq[ks], bfr, acc);
      }
      #pragma unroll
      for (int v = 0; v < 8; ++v)
        S[(rt * 16 + mo + v) * N_ + ct * 16 + ncol] = acc[v] * scale;
    }
  }
  __syncthreads();

  // phase 2: row softmax (one wave per row, 4 rows per wave); rows cached in regs
  for (int i = 0; i < 4; ++i) {
    int row = wave * 4 + i;
    float* Sr = S + (size_t)row * N_;
    float vals[32];
    float mx = -3.4e38f;
    #pragma unroll
    for (int j = 0; j < 32; ++j) { vals[j] = Sr[lane + j * 32]; mx = fmaxf(mx, vals[j]); }
    #pragma unroll
    for (int o = 16; o; o >>= 1) mx = fmaxf(mx, __shfl_xor(mx, o));
    float s = 0.f;
    #pragma unroll
    for (int j = 0; j < 32; ++j) { vals[j] = expf(vals[j] - mx); s += vals[j]; }
    #pragma unroll
    for (int o = 16; o; o >>= 1) s += __shfl_xor(s, o);
    float inv = 1.f / s;
    bf16* Pr = P + (size_t)row * N_;
    #pragma unroll
    for (int j = 0; j < 32; ++j) Pr[lane + j * 32] = (bf16)(vals[j] * inv);
  }
  __syncthreads();

  // phase 3: context = P V ; reduce column sums & maxes; atomics into [B][H]
  {
    v8f acc[2] = {vzero8(), vzero8()};
    for (int ks = 0; ks < 32; ++ks) {
      int m0 = ks * 32;
      v16bf a = load_frag(P + (size_t)(rt * 16 + rr) * N_ + m0 + sel);
      #pragma unroll
      for (int t = 0; t < 2; ++t) {
        int h0 = (cg * 2 + t) * 16;
        v16bf bfr = load_frag(VTb + (size_t)(h0 + rr) * BN + (size_t)b * N_ + m0 + sel);
        acc[t] = wmma_bf(a, bfr, acc[t]);
      }
    }
    #pragma unroll
    for (int t = 0; t < 2; ++t) {
      float s = 0.f, m = -3.4e38f;
      #pragma unroll
      for (int v = 0; v < 8; ++v) { s += acc[t][v]; m = fmaxf(m, acc[t][v]); }
      s += __shfl_xor(s, 16);
      m = fmaxf(m, __shfl_xor(m, 16));
      if (lane < 16) {
        int h = (cg * 2 + t) * 16 + lane;
        atomicAdd(gsum + b * H_ + h, s);
        atomicMaxF(gmax + b * H_ + h, m);
      }
    }
  }
}

// ---------- init / final ----------
__global__ void k_init(float* gsum, float* gmax) {
  int i = blockIdx.x * 256 + threadIdx.x;
  if (i < B_ * H_) { gsum[i] = 0.f; gmax[i] = -3.4e38f; }
}
__global__ void k_final(const float* gsum, const float* gmax, float* out) {
  int i = blockIdx.x * 256 + threadIdx.x;
  if (i < B_ * H_) out[i] = gsum[i] * gmax[i];
}

// ---------- host launcher ----------
extern "C" void kernel_launch(void* const* d_in, const int* in_sizes, int n_in,
                              void* d_out, int out_size, void* d_ws, size_t ws_size,
                              hipStream_t stream) {
  const float* inputs   = (const float*)d_in[0];
  const float* adj      = (const float*)d_in[1];
  const float* mask     = (const float*)d_in[2];
  const float* sa0_w    = (const float*)d_in[3];
  const float* sa0_b    = (const float*)d_in[4];
  const float* ln0_w    = (const float*)d_in[5];
  const float* ln0_b    = (const float*)d_in[6];
  const float* w_rel0   = (const float*)d_in[7];
  const float* w_bases0 = (const float*)d_in[8];
  const float* bias0    = (const float*)d_in[9];
  const float* w_rel1   = (const float*)d_in[10];
  const float* w_bases1 = (const float*)d_in[11];
  const float* bias1    = (const float*)d_in[12];
  const float* q_w      = (const float*)d_in[13];
  const float* q_b      = (const float*)d_in[14];
  const float* k_w      = (const float*)d_in[15];
  const float* k_b      = (const float*)d_in[16];
  const float* v_w      = (const float*)d_in[17];
  const float* v_b      = (const float*)d_in[18];
  float* out = (float*)d_out;

  // workspace carve-up (256B aligned)
  size_t off = 0;
  char* base = (char*)d_ws;
  auto alloc = [&](size_t bytes) -> void* {
    void* p = base + off;
    off += (bytes + 255) & ~(size_t)255;
    return p;
  };
  bf16* adjb = (bf16*)alloc((size_t)B_ * R_ * N_ * N_ * 2);   // 50.3 MB
  bf16* Xb   = (bf16*)alloc((size_t)BN * F_ * 2);
  bf16* WT0  = (bf16*)alloc((size_t)R_ * H_ * F_ * 2);
  bf16* WT1  = (bf16*)alloc((size_t)R_ * H_ * H_ * 2);
  bf16* WTq  = (bf16*)alloc((size_t)H_ * H_ * 2);
  bf16* WTk  = (bf16*)alloc((size_t)H_ * H_ * 2);
  bf16* WTv  = (bf16*)alloc((size_t)H_ * H_ * 2);
  bf16* YTb  = (bf16*)alloc((size_t)R_ * H_ * BN * 2);        // 6 MB
  float* X1  = (float*)alloc((size_t)BN * H_ * 4);
  bf16* X1b  = (bf16*)alloc((size_t)BN * H_ * 2);
  bf16* X2b  = (bf16*)alloc((size_t)BN * H_ * 2);
  bf16* Qb   = (bf16*)alloc((size_t)BN * H_ * 2);
  bf16* Kbf  = (bf16*)alloc((size_t)BN * H_ * 2);
  bf16* VTb  = (bf16*)alloc((size_t)H_ * BN * 2);
  float* gsum = (float*)alloc((size_t)B_ * H_ * 4);
  float* gmax = (float*)alloc((size_t)B_ * H_ * 4);
  (void)ws_size; (void)n_in; (void)in_sizes; (void)out_size;

  // init accumulators
  k_init<<<dim3(4), 256, 0, stream>>>(gsum, gmax);

  // token attention -> Xb
  k_tok<<<dim3(BN), 128, 0, stream>>>(inputs, sa0_w, sa0_b, ln0_w, ln0_b, Xb);

  // adj -> bf16
  {
    int n4 = (B_ * R_ * N_ * N_) / 4;
    k_cast4<<<dim3(n4 / 256), 256, 0, stream>>>(adj, adjb, n4);
  }

  // basis-combined weights (transposed bf16)
  k_basis<<<dim3(64, R_), 256, 0, stream>>>(w_rel0, w_bases0, WT0);
  k_basis<<<dim3(64, R_), 256, 0, stream>>>(w_rel1, w_bases1, WT1);
  k_mat_t<<<dim3(64), 256, 0, stream>>>(q_w, WTq);
  k_mat_t<<<dim3(64), 256, 0, stream>>>(k_w, WTk);
  k_mat_t<<<dim3(64), 256, 0, stream>>>(v_w, WTv);

  // layer 0: Y_r = X W0_r (transposed out), then C = sum_r adj_r Y_r
  for (int r = 0; r < R_; ++r)
    k_gemm128<false, false, true><<<dim3(BN / 128), 256, 0, stream>>>(
        Xb, WT0 + (size_t)r * H_ * F_, nullptr, nullptr,
        YTb + (size_t)r * H_ * BN, BN);
  k_spmm<false><<<dim3(N_ / 32, B_), 256, 0, stream>>>(
      adjb, YTb, bias0, mask, nullptr, X1, X1b);

  // layer 1
  for (int r = 0; r < R_; ++r)
    k_gemm128<false, false, true><<<dim3(BN / 128), 256, 0, stream>>>(
        X1b, WT1 + (size_t)r * H_ * H_, nullptr, nullptr,
        YTb + (size_t)r * H_ * BN, BN);
  k_spmm<true><<<dim3(N_ / 32, B_), 256, 0, stream>>>(
      adjb, YTb, bias1, mask, X1, nullptr, X2b);

  // QKV projections
  k_gemm128<true, true, false><<<dim3(BN / 128), 256, 0, stream>>>(
      X2b, WTq, q_b, Qb, nullptr, BN);
  k_gemm128<true, true, false><<<dim3(BN / 128), 256, 0, stream>>>(
      X2b, WTk, k_b, Kbf, nullptr, BN);
  k_gemm128<true, false, true><<<dim3(BN / 128), 256, 0, stream>>>(
      X2b, WTv, v_b, nullptr, VTb, BN);

  // attention + pooled reduction
  size_t smem_bytes = (size_t)32 * N_ * sizeof(float) + (size_t)32 * N_ * sizeof(bf16);
  k_attn<<<dim3(N_ / 32, B_), 256, smem_bytes, stream>>>(Qb, Kbf, VTb, gsum, gmax);

  // out = sum * max
  k_final<<<dim3(4), 256, 0, stream>>>(gsum, gmax, out);
}
